// GeneTrait_GNN_78073915507270
// MI455X (gfx1250) — matrix operand order repaired and checked by reading, not compiled, not run on hardware
//
#include <hip/hip_runtime.h>

typedef __attribute__((ext_vector_type(16))) __bf16 bf16x16;
typedef __attribute__((ext_vector_type(8)))  float  f32x8;
typedef int v4i __attribute__((vector_size(16)));

#define NODES 50000
#define EDGES 600000
#define PAIRS 200000

// ---------------------------------------------------------------------------
// native f32 global atomic add (no return). Guarantees global_atomic_add_f32
// instead of a possible CAS loop from atomicAdd(float*).
// ---------------------------------------------------------------------------
__device__ inline void gatomic_fadd(float* p, float v) {
#if defined(__gfx1250__)
  asm volatile("global_atomic_add_f32 %0, %1, off"
               :
               : "v"((unsigned long long)(uintptr_t)p), "v"(v)
               : "memory");
#else
  atomicAdd(p, v);
#endif
}

// ---------------------------------------------------------------------------
// async global->LDS 16B copy (ASYNCcnt path) with safe fallback.
// Probe established: param0 is (v4i __device__*) i.e. AS1 pointer to a
// 16-byte vector. Assumed signature: (v4i AS1* src, v4i AS3* dst, int, int).
// ---------------------------------------------------------------------------
#if defined(__HIP_DEVICE_COMPILE__) && __has_builtin(__builtin_amdgcn_global_load_async_to_lds_b128)
#define HAVE_ASYNC_LDS 1
typedef __attribute__((address_space(1))) v4i* gptr_v4i;
typedef __attribute__((address_space(3))) v4i* lptr_v4i;
#else
#define HAVE_ASYNC_LDS 0
#endif

__device__ inline void copy_b128_to_lds(const __bf16* g, __bf16* l) {
#if HAVE_ASYNC_LDS
  __builtin_amdgcn_global_load_async_to_lds_b128(
      (gptr_v4i)(uintptr_t)g, (lptr_v4i)(uintptr_t)l, 0, 0);
#else
  *(ulonglong2*)l = *(const ulonglong2*)g;
#endif
}

__device__ inline void stage_wait() {
#if HAVE_ASYNC_LDS
#if __has_builtin(__builtin_amdgcn_s_wait_asynccnt)
  __builtin_amdgcn_s_wait_asynccnt(0);
#else
  asm volatile("s_wait_asynccnt 0" ::: "memory");
#endif
#endif
}

// ---------------------------------------------------------------------------
// small utility kernels
// ---------------------------------------------------------------------------
__global__ void k_zero_f32(float* __restrict__ p, int n) {
  int i = blockIdx.x * blockDim.x + threadIdx.x;
  if (i < n) p[i] = 0.0f;
}

__global__ void k_degree(float* __restrict__ deg, const int* __restrict__ dst, int e) {
  int i = blockIdx.x * blockDim.x + threadIdx.x;
  if (i < e) gatomic_fadd(&deg[dst[i]], 1.0f);
}

__global__ void k_rsqrt(float* __restrict__ d, int n) {
  int i = blockIdx.x * blockDim.x + threadIdx.x;
  if (i < n) d[i] = rsqrtf(d[i] + 1.0f);   // +1 for self-loop
}

// ---------------------------------------------------------------------------
// pack a [K x 128] row-major f32 weight matrix into WMMA B-fragment layout:
// frag index = c*8 + nt  (c = K-chunk of 32, nt = N-tile of 16)
// within a frag: lane (0..31) * 16 bf16, element e:
//   K = c*32 + (lane<16 ? 0 : 8) + (e&7) + (e>=8 ? 16 : 0),  N = nt*16 + (lane & 15)
// ---------------------------------------------------------------------------
__global__ void k_pack_w128(const float* __restrict__ W, __bf16* __restrict__ Bp, int K) {
  int t = blockIdx.x * blockDim.x + threadIdx.x;
  if (t >= K * 128) return;
  int e    = t & 15;
  int lane = (t >> 4) & 31;
  int frag = t >> 9;
  int nt = frag & 7;
  int c  = frag >> 3;
  int k = c * 32 + ((lane >> 4) << 3) + (e & 7) + ((e >> 3) << 4);
  int n = nt * 16 + (lane & 15);
  Bp[t] = (__bf16)W[k * 128 + n];
}

// ---------------------------------------------------------------------------
// build a 16x32 bf16 A fragment from a row-major f32 row.
// lane<16: elems 0..7 = K koff..koff+7, elems 8..15 = K koff+16..koff+23
// (koff already includes the +8 for lanes>=16)
// ---------------------------------------------------------------------------
__device__ inline bf16x16 make_a_frag(const float* __restrict__ rowp, int koff, int relu) {
  const float4* q0 = (const float4*)(rowp + koff);
  const float4* q1 = (const float4*)(rowp + koff + 16);
  float4 f0 = q0[0], f1 = q0[1], f2 = q1[0], f3 = q1[1];
  float v[16] = {f0.x, f0.y, f0.z, f0.w, f1.x, f1.y, f1.z, f1.w,
                 f2.x, f2.y, f2.z, f2.w, f3.x, f3.y, f3.z, f3.w};
  bf16x16 a;
#pragma unroll
  for (int i = 0; i < 16; i++) {
    float f = v[i];
    if (relu) f = fmaxf(f, 0.0f);
    a[i] = (__bf16)f;
  }
  return a;
}

// ---------------------------------------------------------------------------
// block-cooperative staging of the packed weight matrix (K*128 bf16) into LDS
// via async global->LDS. All 256 threads participate (no early return before
// the barrier).
// ---------------------------------------------------------------------------
__device__ inline void stage_weights(const __bf16* __restrict__ Bp, __bf16* sB, int K) {
  int chunks = K * 16;                       // number of 16-byte chunks
  for (int i = threadIdx.x; i < chunks; i += 256)
    copy_b128_to_lds(Bp + (size_t)i * 8, sB + (size_t)i * 8);
  stage_wait();
  __syncthreads();
}

// ---------------------------------------------------------------------------
// C[M x 128] = A[M x K] @ W[K x 128]   (A f32 row-major, W pre-packed bf16,
// staged in LDS). One wave per 16-row strip; 8 N-tiles -> 8 accumulators.
// Optional fused ReLU on A load (previous layer's activation).
// ---------------------------------------------------------------------------
__global__ __launch_bounds__(256) void k_gemm_n128(
    const float* __restrict__ A, const __bf16* __restrict__ Bp,
    float* __restrict__ C, int Mtiles, int K, int reluIn)
{
  __shared__ __bf16 sB[256 * 128];           // up to 64 KB used
  stage_weights(Bp, sB, K);

  int wave = threadIdx.x >> 5, lane = threadIdx.x & 31;
  int mt = blockIdx.x * 8 + wave;
  if (mt >= Mtiles) return;
  int KC = K >> 5;
  const float* arow = A + (size_t)(mt * 16 + (lane & 15)) * K;
  int klo = (lane < 16) ? 0 : 8;

  f32x8 acc[8] = {};
  for (int c = 0; c < KC; c++) {
    bf16x16 a = make_a_frag(arow, c * 32 + klo, reluIn);
    const __bf16* bbase = sB + (size_t)(c * 8) * 512 + lane * 16;
#pragma unroll
    for (int nt = 0; nt < 8; nt++) {
      bf16x16 b = *(const bf16x16*)(bbase + nt * 512);
      acc[nt] = __builtin_amdgcn_wmma_f32_16x16x32_bf16(
          false, a, false, b, (short)0, acc[nt], false, false);
    }
  }
  // C/D layout: VGPR j -> M = j + (lane>=16 ? 8 : 0), N = lane & 15
  int r0  = mt * 16 + ((lane >> 4) << 3);
  int col = lane & 15;
#pragma unroll
  for (int nt = 0; nt < 8; nt++)
#pragma unroll
    for (int j = 0; j < 8; j++)
      C[(size_t)(r0 + j) * 128 + nt * 16 + col] = acc[nt][j];
}

// ---------------------------------------------------------------------------
// agg[i] = hG[i] * dis[i]^2 + bias   (self-loop term, initializes agg buffer)
// ---------------------------------------------------------------------------
__global__ void k_selfagg(const float* __restrict__ hG, const float* __restrict__ dis,
                          const float* __restrict__ bias, float* __restrict__ agg)
{
  int t = blockIdx.x * blockDim.x + threadIdx.x;   // NODES * 32 (float4 quads)
  if (t >= NODES * 32) return;
  int node = t >> 5, q = t & 31;
  float s = dis[node];
  s = s * s;
  float4 hv = ((const float4*)hG)[t];
  float4 bv = ((const float4*)bias)[q];
  float4 r;
  r.x = hv.x * s + bv.x;
  r.y = hv.y * s + bv.y;
  r.z = hv.z * s + bv.z;
  r.w = hv.w * s + bv.w;
  ((float4*)agg)[t] = r;
}

// ---------------------------------------------------------------------------
// edge scatter: agg[dst] += hG[src] * dis[src]*dis[dst]
// one wave per edge: 32 lanes x float4 = 128 contiguous features.
// ---------------------------------------------------------------------------
__global__ __launch_bounds__(256) void k_edgeagg(
    const float* __restrict__ hG, const float* __restrict__ dis,
    const int* __restrict__ src, const int* __restrict__ dst,
    float* __restrict__ agg)
{
  int gid = blockIdx.x * blockDim.x + threadIdx.x;
  int e = gid >> 5, lane = gid & 31;
  if (e >= EDGES) return;
  int s = src[e], d = dst[e];
  float w = dis[s] * dis[d];
  float4 hv = ((const float4*)(hG + (size_t)s * 128))[lane];
  float* base = agg + (size_t)d * 128 + lane * 4;
  gatomic_fadd(base + 0, hv.x * w);
  gatomic_fadd(base + 1, hv.y * w);
  gatomic_fadd(base + 2, hv.z * w);
  gatomic_fadd(base + 3, hv.w * w);
}

// ---------------------------------------------------------------------------
// link predictor, fully fused:
//   e = concat(h[p0], h[p1])            (gathered A fragments, K=256)
//   z = relu(e @ Wl1 + bl1)             (WMMA + epilogue)
//   out = sigmoid(z @ Wl2 + bl2)        (cross-lane dot in epilogue)
// ---------------------------------------------------------------------------
__global__ __launch_bounds__(256) void k_link(
    const float* __restrict__ Hh, const int* __restrict__ p0, const int* __restrict__ p1,
    const __bf16* __restrict__ Bp, const float* __restrict__ bl1,
    const float* __restrict__ Wl2, const float* __restrict__ bl2,
    float* __restrict__ out, int Mtiles)
{
  __shared__ __bf16 sB[256 * 128];           // 64 KB, Wl1 packed
  stage_weights(Bp, sB, 256);

  int wave = threadIdx.x >> 5, lane = threadIdx.x & 31;
  int mt = blockIdx.x * 8 + wave;
  if (mt >= Mtiles) return;
  int m = mt * 16 + (lane & 15);
  const float* r0 = Hh + (size_t)p0[m] * 128;
  const float* r1 = Hh + (size_t)p1[m] * 128;
  int klo = (lane < 16) ? 0 : 8;

  f32x8 acc[8] = {};
  for (int c = 0; c < 8; c++) {
    const float* rp = (c < 4) ? r0 : r1;          // K<128 -> h[p0], else h[p1]
    bf16x16 a = make_a_frag(rp, (c & 3) * 32 + klo, 0);
    const __bf16* bbase = sB + (size_t)(c * 8) * 512 + lane * 16;
#pragma unroll
    for (int nt = 0; nt < 8; nt++) {
      bf16x16 b = *(const bf16x16*)(bbase + nt * 512);
      acc[nt] = __builtin_amdgcn_wmma_f32_16x16x32_bf16(
          false, a, false, b, (short)0, acc[nt], false, false);
    }
  }
  // epilogue: bias + relu + dot with Wl2 (sum over all 128 N), then sigmoid
  float sum[8] = {0, 0, 0, 0, 0, 0, 0, 0};
#pragma unroll
  for (int nt = 0; nt < 8; nt++) {
    int n = nt * 16 + (lane & 15);
    float bb = bl1[n], wv = Wl2[n];
#pragma unroll
    for (int j = 0; j < 8; j++) {
      float v = acc[nt][j] + bb;
      v = fmaxf(v, 0.0f);
      sum[j] += v * wv;
    }
  }
  // reduce over the 16 lanes holding N=0..15 of the same row
#pragma unroll
  for (int j = 0; j < 8; j++) {
#pragma unroll
    for (int mk = 1; mk < 16; mk <<= 1)
      sum[j] += __shfl_xor(sum[j], mk, 32);
  }
  if ((lane & 15) == 0) {
    int rbase = mt * 16 + ((lane >> 4) << 3);   // lane 0 -> rows M..M+7, lane 16 -> M+8..M+15
    float b2 = bl2[0];
#pragma unroll
    for (int j = 0; j < 8; j++)
      out[rbase + j] = 1.0f / (1.0f + __expf(-(sum[j] + b2)));
  }
}

// ---------------------------------------------------------------------------
extern "C" void kernel_launch(void* const* d_in, const int* in_sizes, int n_in,
                              void* d_out, int out_size, void* d_ws, size_t ws_size,
                              hipStream_t stream) {
  const float* x   = (const float*)d_in[0];
  const int*   ei  = (const int*)d_in[1];
  const int*   ep  = (const int*)d_in[2];
  const float* W0  = (const float*)d_in[3];  const float* b0  = (const float*)d_in[4];
  const float* W1  = (const float*)d_in[5];  const float* b1  = (const float*)d_in[6];
  const float* W2  = (const float*)d_in[7];  const float* b2  = (const float*)d_in[8];
  const float* Wl1 = (const float*)d_in[9];  const float* bl1 = (const float*)d_in[10];
  const float* Wl2 = (const float*)d_in[11]; const float* bl2 = (const float*)d_in[12];
  float* out = (float*)d_out;

  // workspace layout (f32 units): dis | hG | hAgg | packed bf16 weights
  float* ws   = (float*)d_ws;
  float* dis  = ws;                               // 50176 (padded)
  float* hG   = dis + 50176;                      // 50000*128
  float* hAgg = hG + (size_t)NODES * 128;         // 50000*128
  __bf16* W0p  = (__bf16*)(hAgg + (size_t)NODES * 128);
  __bf16* W1p  = W0p + 256 * 128;
  __bf16* W2p  = W1p + 128 * 128;
  __bf16* Wl1p = W2p + 128 * 128;

  const int* esrc = ei;
  const int* edst = ei + EDGES;
  const int* p0 = ep;
  const int* p1 = ep + PAIRS;

  dim3 B(256);

  // degree -> dis = rsqrt(deg+1)
  k_zero_f32<<<dim3((NODES + 255) / 256), B, 0, stream>>>(dis, NODES);
  k_degree  <<<dim3((EDGES + 255) / 256), B, 0, stream>>>(dis, edst, EDGES);
  k_rsqrt   <<<dim3((NODES + 255) / 256), B, 0, stream>>>(dis, NODES);

  // pack weights into WMMA B-fragment layout (bf16)
  k_pack_w128<<<dim3((256 * 128 + 255) / 256), B, 0, stream>>>(W0,  W0p,  256);
  k_pack_w128<<<dim3((128 * 128 + 255) / 256), B, 0, stream>>>(W1,  W1p,  128);
  k_pack_w128<<<dim3((128 * 128 + 255) / 256), B, 0, stream>>>(W2,  W2p,  128);
  k_pack_w128<<<dim3((256 * 128 + 255) / 256), B, 0, stream>>>(Wl1, Wl1p, 256);

  const int Mtiles = NODES / 16;                  // 3125
  const int gemmBlocks = (Mtiles + 7) / 8;
  const int selfBlocks = (NODES * 32 + 255) / 256;
  const int edgeBlocks = EDGES / 8;               // 1 wave per edge

  // layer 0: hG = x @ W0 ; hAgg = aggregate(hG) + b0
  k_gemm_n128<<<dim3(gemmBlocks), B, 0, stream>>>(x, W0p, hG, Mtiles, 256, 0);
  k_selfagg  <<<dim3(selfBlocks), B, 0, stream>>>(hG, dis, b0, hAgg);
  k_edgeagg  <<<dim3(edgeBlocks), B, 0, stream>>>(hG, dis, esrc, edst, hAgg);

  // layer 1 (ReLU of previous output fused into A load)
  k_gemm_n128<<<dim3(gemmBlocks), B, 0, stream>>>(hAgg, W1p, hG, Mtiles, 128, 1);
  k_selfagg  <<<dim3(selfBlocks), B, 0, stream>>>(hG, dis, b1, hAgg);
  k_edgeagg  <<<dim3(edgeBlocks), B, 0, stream>>>(hG, dis, esrc, edst, hAgg);

  // layer 2
  k_gemm_n128<<<dim3(gemmBlocks), B, 0, stream>>>(hAgg, W2p, hG, Mtiles, 128, 1);
  k_selfagg  <<<dim3(selfBlocks), B, 0, stream>>>(hG, dis, b2, hAgg);
  k_edgeagg  <<<dim3(edgeBlocks), B, 0, stream>>>(hG, dis, esrc, edst, hAgg);

  // link predictor (fully fused, writes final sigmoid output)
  const int Ptiles = PAIRS / 16;                  // 12500
  k_link<<<dim3((Ptiles + 7) / 8), B, 0, stream>>>(
      hAgg, p0, p1, Wl1p, bl1, Wl2, bl2, out, Ptiles);
}
